// randlanet_13743895347677
// MI455X (gfx1250) — compile-verified, matrix operand order checked by reading, not run on hardware
//
#include <hip/hip_runtime.h>
#include <hip/hip_fp16.h>

// ---------------------------------------------------------------------------
// RandLA-Net forward for MI455X (gfx1250, wave32, WMMA).
// GEMMs run on v_wmma_f32_16x16x32_f16. Fragment builds are 16B-chunk LDS
// loads (ds_load_b128) exploiting the ISA A-fragment layout: for lane (m,g)
// the 16 halfs are row[8g..8g+7] ++ row[16+8g..16+8g+7] of a row-major tile.
// Epilogue fuses bias + eval-BN + residual add + leaky-relu.
// ---------------------------------------------------------------------------

typedef __attribute__((ext_vector_type(16))) _Float16 v16h;
typedef __attribute__((ext_vector_type(8)))  _Float16 v8h;
typedef __attribute__((ext_vector_type(4)))  _Float16 v4h;
typedef __attribute__((ext_vector_type(8)))  float    v8f;
typedef __attribute__((ext_vector_type(4)))  float    v4f;

#define LRELU(x) ((x) > 0.0f ? (x) : 0.2f * (x))

// ---------------------------------------------------------------------------
// Fused GEMM:  C[M,N] = epi( A[M,K] @ W[K,N] )
// epi: +bias[n] (opt) -> *g[n]/sqrt(1+eps)+be[n] (opt) -> +addsrc[m,n] (opt)
//      -> leaky_relu (opt).  f16 operands, f32 WMMA accumulate.
// Block: 256 thr = 8 waves. Block tile 128x32; each wave owns a 16x32 strip
// (one A fragment, two B fragments, two WMMAs per K-step).
// ---------------------------------------------------------------------------
#define TM 128
#define TN 32
#define TK 32
#define AS 40   // A LDS row stride in halfs (80 B: 16B-aligned, bank-rotating)
#define WS 40   // Wt LDS row stride in halfs

__global__ __launch_bounds__(256, 2) void gemm_wmma_kernel(
    const float* __restrict__ A, const float* __restrict__ W,
    const float* __restrict__ bias, const float* __restrict__ gamma,
    const float* __restrict__ beta, const float* __restrict__ addsrc,
    float* __restrict__ C, int M, int K, int N, int act) {
  __shared__ __align__(16) _Float16 As[TM * AS];   // row-major A tile
  __shared__ __align__(16) _Float16 Wt[TN * WS];   // TRANSPOSED W tile [n][k]

  const int tid  = threadIdx.x;
  const int wave = tid >> 5;
  const int lane = tid & 31;
  const int m0   = blockIdx.x * TM;
  const int n0   = blockIdx.y * TN;
  const bool k4  = (K & 3) == 0;   // uniform: 4-wide aligned A staging legal

  v8f acc0 = {}, acc1 = {};

  for (int k0 = 0; k0 < K; k0 += TK) {
    // Uniform prefetch of the next K-slab of A (one line per tile row).
    if (k0 + TK < K && tid < TM) {
      int gm = m0 + tid;
      if (gm < M) __builtin_prefetch(&A[(size_t)gm * K + k0 + TK], 0, 1);
    }
    // Stage A tile (128x32) row-major as f16, packed ds_store_b64.
    // Fast path (K%4==0): one aligned global_load_b128 per 4-half pack.
    for (int i = tid; i < (TM * TK) / 4; i += 256) {
      int r = i >> 3;            // tile row 0..127
      int c = (i & 7) * 4;       // k offset 0,4,..,28
      int gm = m0 + r;
      v4h pk;
      if (k4) {
        v4f vv = {};
        if (gm < M && (k0 + c) < K)
          vv = *(const v4f*)&A[(size_t)gm * K + k0 + c];
        pk[0] = (_Float16)vv.x; pk[1] = (_Float16)vv.y;
        pk[2] = (_Float16)vv.z; pk[3] = (_Float16)vv.w;
      } else {
#pragma unroll
        for (int j = 0; j < 4; ++j) {
          int gk = k0 + c + j;
          float v = (gm < M && gk < K) ? A[(size_t)gm * K + gk] : 0.0f;
          pk[j] = (_Float16)v;
        }
      }
      *(v4h*)&As[r * AS + c] = pk;
    }
    // Stage W tile transposed: Wt[n][k]. 256 packed elements -> 1 per thread.
    {
      int n = tid >> 3;          // 0..31
      int c = (tid & 7) * 4;     // k offset
      v4h pk;
#pragma unroll
      for (int j = 0; j < 4; ++j) {
        int gk = k0 + c + j, gn = n0 + n;
        float v = (gk < K && gn < N) ? W[(size_t)gk * N + gn] : 0.0f;
        pk[j] = (_Float16)v;
      }
      *(v4h*)&Wt[n * WS + c] = pk;
    }
    __syncthreads();

    // Fragment build: two aligned 16B LDS loads per operand (ISA layout:
    // lane (m, g=lane/16) holds row[8g..8g+7] ++ row[16+8g..16+8g+7]).
    const int mr = wave * 16 + (lane & 15);
    const int nr = lane & 15;
    const int g  = lane >> 4;
    const v8h* ar  = (const v8h*)&As[mr * AS];
    const v8h* br0 = (const v8h*)&Wt[nr * WS];
    const v8h* br1 = (const v8h*)&Wt[(nr + 16) * WS];
    v16h a  = __builtin_shufflevector(ar[g],  ar[2 + g],
                                      0,1,2,3,4,5,6,7,8,9,10,11,12,13,14,15);
    v16h b0 = __builtin_shufflevector(br0[g], br0[2 + g],
                                      0,1,2,3,4,5,6,7,8,9,10,11,12,13,14,15);
    v16h b1 = __builtin_shufflevector(br1[g], br1[2 + g],
                                      0,1,2,3,4,5,6,7,8,9,10,11,12,13,14,15);
    acc0 = __builtin_amdgcn_wmma_f32_16x16x32_f16(false, a, false, b0,
                                                  (short)0, acc0, false, false);
    acc1 = __builtin_amdgcn_wmma_f32_16x16x32_f16(false, a, false, b1,
                                                  (short)0, acc1, false, false);
    __syncthreads();
  }

  // Epilogue. C/D layout: lane holds column n=lane%16; VGPR r holds row
  // m = r + 8*(lane/16) within each 16x16 tile.
  const float bnrs = rsqrtf(1.0f + 1e-6f);
#pragma unroll
  for (int s = 0; s < 2; ++s) {
    const v8f& acc = s ? acc1 : acc0;
    int n = n0 + s * 16 + (lane & 15);
    if (n < N) {
      float bv = bias  ? bias[n]         : 0.0f;
      float sc = gamma ? gamma[n] * bnrs : 1.0f;
      float sh = gamma ? beta[n]         : 0.0f;
#pragma unroll
      for (int r = 0; r < 8; ++r) {
        int m = m0 + wave * 16 + r + 8 * (lane >> 4);
        if (m < M) {
          float x = acc[r] + bv;
          if (gamma)  x = x * sc + sh;
          if (addsrc) x += addsrc[(size_t)m * N + n];
          if (act)    x = LRELU(x);
          C[(size_t)m * N + n] = x;
        }
      }
    }
  }
}

// --------------------------- elementwise kernels ---------------------------

// fc0 (3->8) + bn0 + leaky, one thread per point.
__global__ void fc0_kernel(const float* __restrict__ in,
                           const float* __restrict__ w,
                           const float* __restrict__ b,
                           const float* __restrict__ g,
                           const float* __restrict__ be,
                           float* __restrict__ out, int M) {
  int m = blockIdx.x * blockDim.x + threadIdx.x;
  if (m >= M) return;
  float x0 = in[3 * m], x1 = in[3 * m + 1], x2 = in[3 * m + 2];
  const float s = rsqrtf(1.0f + 1e-6f);
#pragma unroll
  for (int c = 0; c < 8; ++c) {
    float y = x0 * w[c] + x1 * w[8 + c] + x2 * w[16 + c] + b[c];
    y = y * s * g[c] + be[c];
    out[(size_t)m * 8 + c] = LRELU(y);
  }
}

// Relative position encoding: (B,N,K) -> 10 features [dist, rel, tile, nb].
__global__ void relpos_kernel(const float* __restrict__ xyz,
                              const int* __restrict__ idx,
                              float* __restrict__ out, int N, long total) {
  long t = (long)blockIdx.x * blockDim.x + threadIdx.x;
  if (t >= total) return;
  long nk = t / 16;              // (b*N+n)
  int  b  = (int)(nk / N);
  int  n  = (int)(nk % N);
  const float* pc = xyz + ((size_t)b * N + n) * 3;
  int j = idx[t];
  const float* pn = xyz + ((size_t)b * N + j) * 3;
  float rx = pc[0] - pn[0], ry = pc[1] - pn[1], rz = pc[2] - pn[2];
  float d = sqrtf(rx * rx + ry * ry + rz * rz);
  float* o = out + (size_t)t * 10;
  o[0] = d;  o[1] = rx; o[2] = ry; o[3] = rz;
  o[4] = pc[0]; o[5] = pc[1]; o[6] = pc[2];
  o[7] = pn[0]; o[8] = pn[1]; o[9] = pn[2];
}

// out[(b,n,k), 0:dh] = fpc[b, neigh[b,n,k], :] ; out[(b,n,k), dh:2dh] = fxyz row
__global__ void gather_concat_kernel(const float* __restrict__ fpc,
                                     const float* __restrict__ fxyz,
                                     const int* __restrict__ idx,
                                     float* __restrict__ out,
                                     int N, int dh, long total) {
  long t = (long)blockIdx.x * blockDim.x + threadIdx.x;
  if (t >= total) return;
  int d = 2 * dh;
  long row = t / d;
  int  c   = (int)(t % d);
  if (c < dh) {
    int b = (int)(row / ((long)N * 16));
    int j = idx[row];
    out[t] = fpc[((size_t)b * N + j) * dh + c];
  } else {
    out[t] = fxyz[row * (size_t)dh + (c - dh)];
  }
}

// Attentive pooling: softmax over K=16 of S, weighted sum of F. Thread=(m,c).
__global__ void attpool_kernel(const float* __restrict__ S,
                               const float* __restrict__ F,
                               float* __restrict__ out, int d, long Mtot) {
  long t = (long)blockIdx.x * blockDim.x + threadIdx.x;
  if (t >= Mtot * (long)d) return;
  long m = t / d;
  int  c = (int)(t % d);
  const float* sp = S + (size_t)(m * 16) * d + c;
  const float* fp = F + (size_t)(m * 16) * d + c;
  float mx = -3.4e38f;
#pragma unroll
  for (int k = 0; k < 16; ++k) mx = fmaxf(mx, sp[(size_t)k * d]);
  float sum = 0.0f, agg = 0.0f;
#pragma unroll
  for (int k = 0; k < 16; ++k) {
    float e = __expf(sp[(size_t)k * d] - mx);
    sum += e;
    agg += e * fp[(size_t)k * d];
  }
  out[(size_t)m * d + c] = agg / sum;
}

// Subsample: max over K=16 gathered neighbors. Thread=(b,n_out,c).
__global__ void maxpool_kernel(const float* __restrict__ f,
                               const int* __restrict__ sub,
                               float* __restrict__ out,
                               int Nin, int Nout, int d, long total) {
  long t = (long)blockIdx.x * blockDim.x + threadIdx.x;
  if (t >= total) return;
  int  c  = (int)(t % d);
  long mo = t / d;
  int  b  = (int)(mo / Nout);
  int  n  = (int)(mo % Nout);
  const int* sp = sub + ((size_t)b * Nout + n) * 16;
  float mx = -3.4e38f;
#pragma unroll
  for (int k = 0; k < 16; ++k)
    mx = fmaxf(mx, f[((size_t)b * Nin + sp[k]) * d + c]);
  out[t] = mx;
}

// Nearest-neighbor upsample. Thread=(b,n_out,c).
__global__ void interp_kernel(const float* __restrict__ f,
                              const int* __restrict__ idx,
                              float* __restrict__ out,
                              int Nin, int Nout, int d, long total) {
  long t = (long)blockIdx.x * blockDim.x + threadIdx.x;
  if (t >= total) return;
  int  c  = (int)(t % d);
  long mo = t / d;
  int  b  = (int)(mo / Nout);
  int  n  = (int)(mo % Nout);
  int  j  = idx[(size_t)b * Nout + n];
  out[t] = f[((size_t)b * Nin + j) * d + c];
}

// Channel concat.
__global__ void concat_kernel(const float* __restrict__ A,
                              const float* __restrict__ Bp,
                              float* __restrict__ out,
                              int d1, int d2, long rows) {
  int d = d1 + d2;
  long t = (long)blockIdx.x * blockDim.x + threadIdx.x;
  if (t >= rows * (long)d) return;
  long r = t / d;
  int  c = (int)(t % d);
  out[t] = (c < d1) ? A[r * (size_t)d1 + c] : Bp[r * (size_t)d2 + (c - d1)];
}

// fc3 (32->19) + transpose to (B, 19, N). Thread per point.
__global__ void fc3_kernel(const float* __restrict__ in,
                           const float* __restrict__ w,
                           const float* __restrict__ b,
                           float* __restrict__ out, int Npts, long total) {
  long m = (long)blockIdx.x * blockDim.x + threadIdx.x;
  if (m >= total) return;
  int bb = (int)(m / Npts);
  int n  = (int)(m % Npts);
  const float* x = in + (size_t)m * 32;
#pragma unroll
  for (int c = 0; c < 19; ++c) {
    float y = b[c];
#pragma unroll
    for (int i = 0; i < 32; ++i) y += x[i] * w[i * 19 + c];
    out[((size_t)bb * 19 + c) * Npts + n] = y;
  }
}

// ------------------------------ host driver --------------------------------

struct Conv { const float *b, *be, *g, *w; };

static inline void gemm(hipStream_t st, const float* A, const float* W,
                        const float* b, const float* g, const float* be,
                        const float* add, float* C, long M, int K, int N,
                        int act) {
  dim3 grid((unsigned)((M + TM - 1) / TM), (unsigned)((N + TN - 1) / TN));
  gemm_wmma_kernel<<<grid, 256, 0, st>>>(A, W, b, g, be, add, C,
                                         (int)M, K, N, act);
}
static inline unsigned gblk(long total) {
  return (unsigned)((total + 255) / 256);
}

extern "C" void kernel_launch(void* const* d_in, const int* in_sizes, int n_in,
                              void* d_out, int out_size, void* d_ws,
                              size_t ws_size, hipStream_t stream) {
  (void)in_sizes; (void)n_in; (void)out_size; (void)ws_size;

  static const int  NSv[5]  = {45056, 11264, 2816, 704, 176};
  static const int  DOUT[4] = {16, 64, 128, 256};
  const int Bv = 2;
  const long M0 = (long)Bv * NSv[0];

  // -- Unpack d_in. Order: features, params (pytree-flattened, dict keys in
  //    alphabetical order at each level, lists in index order), then
  //    xyz{i}, neigh{i}, sub{i}, interp{i} for i=0..3.
  int pi = 0;
  const float* features = (const float*)d_in[pi++];
  auto nextf = [&]() { return (const float*)d_in[pi++]; };
  auto conv  = [&]() { Conv c; c.b = nextf(); c.be = nextf();
                       c.g = nextf(); c.w = nextf(); return c; };

  const float* bn0_be = nextf();             // 'bn0': {be, g}
  const float* bn0_g  = nextf();
  Conv dec_[4];
  for (int j = 0; j < 4; ++j) dec_[j] = conv();   // 'dec' list
  Conv dec0c = conv();                            // 'dec0'
  struct EncP { Conv att1m; const float* att1wa; Conv att2m;
                const float* att2wa; Conv mlp1, mlp2, shortcut, xyz1, xyz2; };
  EncP E[4];
  for (int i = 0; i < 4; ++i) {                   // 'enc' list
    E[i].att1m = conv(); E[i].att1wa = nextf();   //  att1: {mlp, wa}
    E[i].att2m = conv(); E[i].att2wa = nextf();   //  att2
    E[i].mlp1 = conv(); E[i].mlp2 = conv();
    E[i].shortcut = conv(); E[i].xyz1 = conv(); E[i].xyz2 = conv();
  }
  const float* fc0_b = nextf(); const float* fc0_w = nextf();  // 'fc0': {b,w}
  Conv fc1c = conv();                                          // 'fc1'
  Conv fc2c = conv();                                          // 'fc2'
  const float* fc3_b = nextf(); const float* fc3_w = nextf();  // 'fc3': {b,w}

  const float* xyzp[4]; const int* neigh[4]; const int* sub[4];
  const int* itp[4];
  for (int i = 0; i < 4; ++i) {
    xyzp[i]  = (const float*)d_in[pi++];
    neigh[i] = (const int*)d_in[pi++];
    sub[i]   = (const int*)d_in[pi++];
    itp[i]   = (const int*)d_in[pi++];
  }

  // -- Workspace carve (bump allocator; peak ~440 MB of scratch).
  float* wsf = (float*)d_ws;
  size_t off = 0;
  auto alloc = [&](size_t n) { float* p = wsf + off; off += n; return p; };

  float* feat0 = alloc((size_t)M0 * 8);        // fc0 output
  float* fe0   = alloc((size_t)M0 * 32);       // enc[0] (persists to decoder)
  float* fl1   = alloc((size_t)Bv * NSv[1] * 32);
  float* fl2   = alloc((size_t)Bv * NSv[2] * 128);
  float* fl3   = alloc((size_t)Bv * NSv[3] * 256);
  float* fl4   = alloc((size_t)Bv * NSv[4] * 512);
  float* rel   = alloc((size_t)M0 * 16 * 10);  // relpos (reused as f_xyz2)
  float* fxyz  = alloc((size_t)M0 * 16 * 8);   // = max over levels of MK*dh
  float* cat   = alloc((size_t)M0 * 16 * 16);  // = max over levels of MK*d
  float* sco   = alloc((size_t)M0 * 16 * 16);
  float* fpc   = alloc((size_t)M0 * 8);
  float* agg   = alloc((size_t)M0 * 16);
  float* fpc3  = alloc((size_t)M0 * 16);
  float* tsc   = alloc((size_t)M0 * 32);       // shortcut
  float* feS   = alloc((size_t)M0 * 32);       // fe scratch, levels 1..3
  float* dA    = alloc((size_t)M0 * 32);       // decoder ping
  float* dB    = alloc((size_t)M0 * 32);       // decoder pong
  float* fi    = alloc((size_t)M0 * 32);       // interp output
  float* catd  = alloc((size_t)M0 * 64);       // decoder concat
  float* t64   = alloc((size_t)M0 * 64);       // fc1 out
  float* t32   = alloc((size_t)M0 * 32);       // fc2 out

  // -- fc0 + bn0 + leaky
  fc0_kernel<<<gblk(M0), 256, 0, stream>>>(features, fc0_w, fc0_b, bn0_g,
                                           bn0_be, feat0, (int)M0);

  // -- Encoder
  const float* fin = feat0;
  float* flv[5] = {nullptr, fl1, fl2, fl3, fl4};
  int din = 8;
  for (int i = 0; i < 4; ++i) {
    const int  N  = NSv[i];
    const long M  = (long)Bv * N;
    const long MK = M * 16;
    const int  d  = DOUT[i], dh = d / 2;
    const EncP& e = E[i];
    float* fe = (i == 0) ? fe0 : feS;

    // f_pc = conv(feat, mlp1)
    gemm(stream, fin, e.mlp1.w, e.mlp1.b, e.mlp1.g, e.mlp1.be, nullptr,
         fpc, M, din, dh, 1);
    // relpos -> f_xyz = conv(relpos, xyz1)
    relpos_kernel<<<gblk(MK), 256, 0, stream>>>(xyzp[i], neigh[i], rel, N, MK);
    gemm(stream, rel, e.xyz1.w, e.xyz1.b, e.xyz1.g, e.xyz1.be, nullptr,
         fxyz, MK, 10, dh, 1);
    // cat1 = [gather(f_pc), f_xyz]; att-pool 1
    gather_concat_kernel<<<gblk(MK * d), 256, 0, stream>>>(fpc, fxyz, neigh[i],
                                                           cat, N, dh, MK * d);
    gemm(stream, cat, e.att1wa, nullptr, nullptr, nullptr, nullptr,
         sco, MK, d, d, 0);
    attpool_kernel<<<gblk(M * d), 256, 0, stream>>>(sco, cat, agg, d, M);
    gemm(stream, agg, e.att1m.w, e.att1m.b, e.att1m.g, e.att1m.be, nullptr,
         fpc, M, d, dh, 1);
    // f_xyz2 = conv(f_xyz, xyz2)  (rel slot reused)
    float* fxyz2 = rel;
    gemm(stream, fxyz, e.xyz2.w, e.xyz2.b, e.xyz2.g, e.xyz2.be, nullptr,
         fxyz2, MK, dh, dh, 1);
    // cat2; att-pool 2
    gather_concat_kernel<<<gblk(MK * d), 256, 0, stream>>>(fpc, fxyz2,
                                                           neigh[i], cat, N,
                                                           dh, MK * d);
    gemm(stream, cat, e.att2wa, nullptr, nullptr, nullptr, nullptr,
         sco, MK, d, d, 0);
    attpool_kernel<<<gblk(M * d), 256, 0, stream>>>(sco, cat, agg, d, M);
    gemm(stream, agg, e.att2m.w, e.att2m.b, e.att2m.g, e.att2m.be, nullptr,
         fpc3, M, d, d, 1);
    // shortcut (bn, no act) then mlp2 (bn) + add + leaky, fused epilogue
    gemm(stream, fin, e.shortcut.w, e.shortcut.b, e.shortcut.g, e.shortcut.be,
         nullptr, tsc, M, din, 2 * d, 0);
    gemm(stream, fpc3, e.mlp2.w, e.mlp2.b, e.mlp2.g, e.mlp2.be, tsc,
         fe, M, d, 2 * d, 1);
    // subsample (max over K gathered neighbors)
    const long Mn = (long)Bv * NSv[i + 1];
    maxpool_kernel<<<gblk(Mn * 2 * d), 256, 0, stream>>>(
        fe, sub[i], flv[i + 1], N, NSv[i + 1], 2 * d, Mn * 2 * d);
    fin = flv[i + 1];
    din = 2 * d;
  }

  // -- Decoder
  float* f = dA;
  gemm(stream, fl4, dec0c.w, dec0c.b, dec0c.g, dec0c.be, nullptr,
       f, (long)Bv * NSv[4], 512, 512, 1);
  int dfeat = 512;
  const float* skipF[4] = {fe0, fl1, fl2, fl3};
  const int    skipC[4] = {32, 32, 128, 256};
  for (int j = 0; j < 4; ++j) {
    const int lev  = 3 - j;
    const int Nout = NSv[lev], Nin = NSv[lev + 1];
    const long Mo  = (long)Bv * Nout;
    interp_kernel<<<gblk(Mo * dfeat), 256, 0, stream>>>(
        f, itp[lev], fi, Nin, Nout, dfeat, Mo * dfeat);
    const int ds = skipC[lev];
    concat_kernel<<<gblk(Mo * (ds + dfeat)), 256, 0, stream>>>(
        skipF[lev], fi, catd, ds, dfeat, Mo);
    float* fn = (f == dA) ? dB : dA;
    gemm(stream, catd, dec_[j].w, dec_[j].b, dec_[j].g, dec_[j].be, nullptr,
         fn, Mo, ds + dfeat, ds, 1);
    f = fn;
    dfeat = ds;
  }

  // -- Heads: fc1 (32->64), fc2 (64->32), fc3 (32->19) + transpose.
  gemm(stream, f,   fc1c.w, fc1c.b, fc1c.g, fc1c.be, nullptr, t64, M0, 32, 64, 1);
  gemm(stream, t64, fc2c.w, fc2c.b, fc2c.g, fc2c.be, nullptr, t32, M0, 64, 32, 1);
  fc3_kernel<<<gblk(M0), 256, 0, stream>>>(t32, fc3_w, fc3_b, (float*)d_out,
                                           NSv[0], M0);
}
